// GroupedExpertLinearAdapter_66305705115884
// MI455X (gfx1250) — matrix operand order repaired and checked by reading, not compiled, not run on hardware
//
#include <hip/hip_runtime.h>

// ---------------------------------------------------------------------------
// Grouped LoRA (16 experts): out = ((X @ A_e) @ B_e) * 0.25, fused bf16 WMMA.
// Software-pipelined: global->LDS staging, LDS->WMMA operand ping-pong, and
// stage-2 global operand ping-pong all overlap memory latency with WMMA.
// ---------------------------------------------------------------------------

typedef __attribute__((ext_vector_type(16))) __bf16 v16bf;
typedef __attribute__((ext_vector_type(8)))  float  v8f;

#define NUM_EXPERTS 16
#define HF 2048            // IN_FEATURES
#define OF 2048            // OUT_FEATURES
#define RANK 128
#define TOKENS 32768
#define TE (TOKENS / NUM_EXPERTS)   // 2048 tokens per expert
#define LORA_SCALE 0.25f            // ALPHA / RANK

union V16 {
  v16bf v;
  uint4 q[2];
  unsigned u[8];
};
union V8F {
  v8f    v;
  float4 q[2];
  float  f[8];
};

__device__ __forceinline__ unsigned pack_bf16(float a, float b) {
  union { __bf16 h[2]; unsigned u; } t;
  t.h[0] = (__bf16)a;
  t.h[1] = (__bf16)b;
  return t.u;
}

// ---------------------------------------------------------------------------
// Prep: per-expert transpose + f32->bf16 convert.
// src: [E][rows][cols] f32   ->   dst: [E][cols][rows] bf16
// ---------------------------------------------------------------------------
__global__ void __launch_bounds__(256)
xpose_cvt_kernel(const float* __restrict__ src, __bf16* __restrict__ dst,
                 int rows, int cols) {
  __shared__ float tile[32][33];
  const int e = blockIdx.z;
  const float* s = src + (size_t)e * rows * cols;
  __bf16*      d = dst + (size_t)e * rows * cols;
  const int c0 = blockIdx.x * 32, r0 = blockIdx.y * 32;
  const int tx = threadIdx.x, ty = threadIdx.y;  // 32 x 8
#pragma unroll
  for (int j = 0; j < 4; ++j)
    tile[ty + j * 8][tx] = s[(size_t)(r0 + ty + j * 8) * cols + (c0 + tx)];
  __syncthreads();
#pragma unroll
  for (int j = 0; j < 4; ++j)
    d[(size_t)(c0 + ty + j * 8) * rows + (r0 + tx)] = (__bf16)tile[tx][ty + j * 8];
}

// ---------------------------------------------------------------------------
// Fused grouped-LoRA kernel.
//   Stage 1 (per wave, 16 tokens):  low^T[128r x 16m] = A^T * X^T   (512 WMMA)
//   Register repack (half-wave shuffles): low^T D-tiles -> stage-2 B operands
//   Stage 2:                        out^T[2048o x 16m] = B^T * low  (512 WMMA)
// Grid: 16 experts * 16 row-blocks = 256 blocks of 256 threads (8 waves).
// ---------------------------------------------------------------------------
#define LDS_ROW_BYTES 80   // 32 bf16 (64B) + 16B pad -> conflict-free b128 reads

__global__ void __launch_bounds__(256)
grouped_lora_kernel(const float* __restrict__ X,
                    const __bf16* __restrict__ AT,   // [E][RANK][HF]
                    const __bf16* __restrict__ BT,   // [E][OF][RANK]
                    float* __restrict__ out) {
  __shared__ __align__(16) unsigned char ldsA[2][128 * LDS_ROW_BYTES];

  const int tid    = threadIdx.x;
  const int lane   = tid & 31;
  const int wave   = tid >> 5;
  const int lane16 = lane & 15;
  const int hi     = lane >> 4;          // which half of the wave

  const int e  = blockIdx.x >> 4;        // expert
  const int mb = blockIdx.x & 15;        // 128-token block within expert
  const int m0 = e * TE + mb * 128 + wave * 16;   // wave's first global token

  const float*  Xrow = X  + (size_t)(m0 + lane16) * HF;
  const __bf16* Ae   = AT + (size_t)e * RANK * HF;  // [r][k], k contiguous
  const __bf16* Be   = BT + (size_t)e * OF * RANK;  // [o][r], r contiguous

  // ---------------- Stage 1: low^T = A^T * X^T ----------------
  v8f acc[8];
  const v8f zero = {0.f, 0.f, 0.f, 0.f, 0.f, 0.f, 0.f, 0.f};
#pragma unroll
  for (int t = 0; t < 8; ++t) acc[t] = zero;

  // cooperative copy: 2 threads per r-row, 32B each (128 rows x 64B per chunk)
  const int crow  = tid >> 1;
  const int cpart = (tid & 1) * 32;
  const unsigned char* gArow = (const unsigned char*)(Ae + (size_t)crow * HF) + cpart;
  uint4* lrow = (uint4*)(&ldsA[0][0] + crow * LDS_ROW_BYTES + cpart);

  // prime the global->LDS pipeline with chunk 0
  uint4 g0, g1;
  {
    const uint4* g = (const uint4*)(gArow);
    g0 = g[0];
    g1 = g[1];
  }

  // LDS read helper offsets (per wave)
  const unsigned ldsRdBase = (unsigned)(lane16 * LDS_ROW_BYTES + hi * 16);

  for (int kc = 0; kc < HF / 32; ++kc) {
    const int k0 = kc * 32;
    const unsigned bufOff = (unsigned)((kc & 1) * (128 * LDS_ROW_BYTES));
    // commit staged chunk kc to LDS
    {
      uint4* l = (uint4*)((unsigned char*)lrow + bufOff);
      l[0] = g0;
      l[1] = g1;
    }
    // issue global loads for chunk kc+1 (overlaps with this chunk's WMMAs)
    if (kc + 1 < HF / 32) {
      const uint4* g = (const uint4*)(gArow + (size_t)(k0 + 32) * 2);
      g0 = g[0];
      g1 = g[1];
    }
    // B operand: X^T chunk, lane = token column; f32 at k 8*hi..+7 and 16+8*hi..+7 -> bf16
    V16 xb;
    {
      const float4* xp = (const float4*)(Xrow + k0 + hi * 8);
      float4 f0 = xp[0], f1 = xp[1], f2 = xp[4], f3 = xp[5];
      __builtin_prefetch(Xrow + k0 + 32, 0, 1);   // global_prefetch next chunk
      xb.u[0] = pack_bf16(f0.x, f0.y); xb.u[1] = pack_bf16(f0.z, f0.w);
      xb.u[2] = pack_bf16(f1.x, f1.y); xb.u[3] = pack_bf16(f1.z, f1.w);
      xb.u[4] = pack_bf16(f2.x, f2.y); xb.u[5] = pack_bf16(f2.z, f2.w);
      xb.u[6] = pack_bf16(f3.x, f3.y); xb.u[7] = pack_bf16(f3.z, f3.w);
    }
    __syncthreads();   // LDS chunk visible (double-buffered: one barrier/iter)

    const unsigned char* rd = &ldsA[0][0] + bufOff + ldsRdBase;
    // ping-pong A-operand registers: next tile's DS loads in flight during WMMA
    V16 a0, a1;
    a0.q[0] = *(const uint4*)(rd);
    a0.q[1] = *(const uint4*)(rd + 32);
#pragma unroll
    for (int t = 0; t < 8; t += 2) {
      const unsigned char* r1 = rd + (t + 1) * (16 * LDS_ROW_BYTES);
      a1.q[0] = *(const uint4*)(r1);
      a1.q[1] = *(const uint4*)(r1 + 32);
      acc[t] = __builtin_amdgcn_wmma_f32_16x16x32_bf16(
          false, a0.v, false, xb.v, (short)0, acc[t], false, false);
      if (t + 2 < 8) {
        const unsigned char* r2 = rd + (t + 2) * (16 * LDS_ROW_BYTES);
        a0.q[0] = *(const uint4*)(r2);
        a0.q[1] = *(const uint4*)(r2 + 32);
      }
      acc[t + 1] = __builtin_amdgcn_wmma_f32_16x16x32_bf16(
          false, a1.v, false, xb.v, (short)0, acc[t + 1], false, false);
    }
  }

  // ------- Repack low^T D-tiles into stage-2 B operands (registers only) ----
  // D-tile t, VGPR p: lanes<16 -> row r=16t+p, lanes>=16 -> r=16t+8+p, col m=lane16.
  // B-operand chunk c, VGPR v: lanes<16 rows 32c+2v,+1 ; lanes>=16 rows 32c+16+2v,+1.
  V16 bop[4];
#pragma unroll
  for (int c = 0; c < 4; ++c) {
#pragma unroll
    for (int v = 0; v < 8; ++v) {
      const int p = (v < 4) ? (2 * v) : (2 * v - 8);
      unsigned from2c  = pack_bf16(acc[2 * c    ][p] * LORA_SCALE,
                                   acc[2 * c    ][p + 1] * LORA_SCALE);
      unsigned from2c1 = pack_bf16(acc[2 * c + 1][p] * LORA_SCALE,
                                   acc[2 * c + 1][p + 1] * LORA_SCALE);
      unsigned val;
      if (v < 4) {
        // low half: own tile 2c; high half: tile 2c+1 data lives in lanes<16
        unsigned sw = (unsigned)__shfl_xor((int)from2c1, 16, 32);
        val = hi ? sw : from2c;
      } else {
        // low half: tile 2c data lives in lanes>=16; high half: own tile 2c+1
        unsigned sw = (unsigned)__shfl_xor((int)from2c, 16, 32);
        val = hi ? from2c1 : sw;
      }
      bop[c].u[v] = val;
    }
  }

  // ---------------- Stage 2: out^T = B^T * low ----------------
  float* orow = out + (size_t)(m0 + lane16) * OF;
  const __bf16* bpBase = Be + (size_t)lane16 * RANK;

  // A-operand loader for one 16-wide o-tile (4 k-chunks, 8x b128)
  auto loadA = [&](V16* dst, int ot) {
    const __bf16* bp = bpBase + (size_t)(ot * 16) * RANK;
#pragma unroll
    for (int c = 0; c < 4; ++c) {
      dst[c].q[0] = *(const uint4*)(bp + 32 * c + hi * 16);
      dst[c].q[1] = *(const uint4*)(bp + 32 * c + hi * 16 + 8);
    }
  };
  auto computeStore = [&](const V16* a, int ot) {
    v8f d = zero;
#pragma unroll
    for (int c = 0; c < 4; ++c)
      d = __builtin_amdgcn_wmma_f32_16x16x32_bf16(
          false, a[c].v, false, bop[c].v, (short)0, d, false, false);
    // D: VGPR p -> o = ot*16 + 8*hi + p, token = m0+lane16. 32B contiguous/lane.
    V8F dv; dv.v = d;
    float4* op = (float4*)(orow + ot * 16 + hi * 8);
    op[0] = dv.q[0];
    op[1] = dv.q[1];
  };

  V16 pa[4], pb[4];
  loadA(pa, 0);
  for (int ot = 0; ot < OF / 16; ot += 2) {
    loadA(pb, ot + 1);          // in flight during computeStore(pa)
    computeStore(pa, ot);
    if (ot + 2 < OF / 16) loadA(pa, ot + 2);  // in flight during computeStore(pb)
    computeStore(pb, ot + 1);
  }
}

// ---------------------------------------------------------------------------
extern "C" void kernel_launch(void* const* d_in, const int* in_sizes, int n_in,
                              void* d_out, int out_size, void* d_ws, size_t ws_size,
                              hipStream_t stream) {
  const float* x      = (const float*)d_in[0];
  const float* lora_a = (const float*)d_in[1];   // [E][HF][RANK]
  const float* lora_b = (const float*)d_in[2];   // [E][RANK][OF]
  float* out = (float*)d_out;

  __bf16* AT = (__bf16*)d_ws;                            // [E][RANK][HF] bf16, 8 MB
  __bf16* BT = AT + (size_t)NUM_EXPERTS * RANK * HF;     // [E][OF][RANK] bf16, 8 MB

  dim3 tb(32, 8);
  // lora_a [E][HF][RANK] -> AT [E][RANK][HF]
  xpose_cvt_kernel<<<dim3(RANK / 32, HF / 32, NUM_EXPERTS), tb, 0, stream>>>(
      lora_a, AT, HF, RANK);
  // lora_b [E][RANK][OF] -> BT [E][OF][RANK]
  xpose_cvt_kernel<<<dim3(OF / 32, RANK / 32, NUM_EXPERTS), tb, 0, stream>>>(
      lora_b, BT, RANK, OF);

  grouped_lora_kernel<<<NUM_EXPERTS * (TE / 128), 256, 0, stream>>>(x, AT, BT, out);
}
